// MultiheadAttention_26036091748848
// MI455X (gfx1250) — matrix operand order repaired and checked by reading, not compiled, not run on hardware
//
#include <hip/hip_runtime.h>
#include <math.h>

// MI455X / gfx1250: wave32, WMMA bf16 16x16x32 with f32 accumulate.
//
// Pipeline (all bf16-staged through d_ws, ~92MB):
//   [0] cvt_kernel  : one-shot fp32 -> bf16 conversion of x and all weights
//                     (keeps v_cvt_pk_bf16_f32 OUT of the GEMM hot loops)
//   [1] qkv_kernel  : x[b,c] @ Wq/Wk/Wv[h,c]^T -> Q,K (row-major bf16), V^T (col-major bf16)
//   [2] attn_kernel : flash attention per (b,h,c), online softmax, 16-row tiles per wave,
//                     probs D->A relayout via per-wave LDS tile (in-order DS, no barrier)
//   [3] outp_kernel : cat[b,c] @ Wo[c]^T -> f32 d_out

typedef __attribute__((ext_vector_type(16))) __bf16 v16bf;
typedef __attribute__((ext_vector_type(8)))  __bf16 v8bf;
typedef __attribute__((ext_vector_type(8)))  float  v8f;
typedef __attribute__((ext_vector_type(4)))  float  v4f;

#define DEVINL __device__ __forceinline__

constexpr int Bn = 4, Hn = 8, Cn = 4, Pn = 1024, En = 512, HDn = 64;

DEVINL v8f v8f_zero() {
  v8f z;
#pragma unroll
  for (int i = 0; i < 8; ++i) z[i] = 0.0f;
  return z;
}

DEVINL v8f wmma_bf16(v16bf a, v16bf b, v8f c) {
  // 8 args: (neg_a, A, neg_b, B, c_mod, C, reuse_a, reuse_b)
  return __builtin_amdgcn_wmma_f32_16x16x32_bf16(false, a, false, b, (short)0, c, false, false);
}

// ---- A operand (16x32, MxK). ISA layout:
//   lane L(0..15):  row L, K = {k0+0..7, k0+16..23}
//   lane L+16:      row L, K = {k0+8..15, k0+24..31}
DEVINL v16bf load_a_bf16(const __bf16* __restrict__ src, int ld, int row0, int k0, int lane) {
  const __bf16* p = src + (size_t)(row0 + (lane & 15)) * ld + k0 + (lane >> 4) * 8;
  v8bf lo = *(const v8bf*)p;
  v8bf hi = *(const v8bf*)(p + 16);
  v16bf a;
#pragma unroll
  for (int i = 0; i < 8; ++i) { a[i] = lo[i]; a[8 + i] = hi[i]; }
  return a;
}

// ---- B operand (32x16, KxN) from B^T storage (row n holds K contiguous):
//   lane n(0..15):  col n, K = {k0+0..7, k0+16..23} ; lane n+16: col n, K = {k0+8..15, k0+24..31}
DEVINL v16bf load_b_bf16(const __bf16* __restrict__ srcNK, int ld, int n0, int k0, int lane) {
  const __bf16* p = srcNK + (size_t)(n0 + (lane & 15)) * ld + k0 + (lane >> 4) * 8;
  v8bf lo = *(const v8bf*)p;
  v8bf hi = *(const v8bf*)(p + 16);
  v16bf b;
#pragma unroll
  for (int i = 0; i < 8; ++i) { b[i] = lo[i]; b[8 + i] = hi[i]; }
  return b;
}

// ======================= [0] fp32 -> bf16 conversion =======================
// Each thread converts 8 floats -> one 16B bf16 store.
__global__ __launch_bounds__(256) void cvt_kernel(const float* __restrict__ src,
                                                  __bf16* __restrict__ dst, int n8) {
  int i = blockIdx.x * 256 + threadIdx.x;
  if (i >= n8) return;
  const v4f* s = (const v4f*)src + (size_t)i * 2;
  v4f f0 = s[0], f1 = s[1];
  v8bf o;
#pragma unroll
  for (int j = 0; j < 4; ++j) { o[j] = (__bf16)f0[j]; o[4 + j] = (__bf16)f1[j]; }
  *((v8bf*)dst + i) = o;
}

// ======================= [1] QKV projection =======================
// grid: B*C*H*(P/128)=1024 blocks x 256 thr (8 waves); wave -> 16 rows, full HD=64.
__global__ __launch_bounds__(256) void qkv_kernel(
    const __bf16* __restrict__ xb, const __bf16* __restrict__ wq,
    const __bf16* __restrict__ wk, const __bf16* __restrict__ wv,
    __bf16* __restrict__ Qb, __bf16* __restrict__ Kb, __bf16* __restrict__ Vt) {
  int blk = blockIdx.x;
  int ptile = blk & 7, h = (blk >> 3) & 7, c = (blk >> 6) & 3, b = blk >> 8;
  int wave = threadIdx.x >> 5, lane = threadIdx.x & 31;
  int p0 = ptile * 128 + wave * 16;

  const __bf16* xbc = xb + (size_t)(b * Cn + c) * Pn * En;   // [P,E]
  const __bf16* Wq  = wq + (size_t)(h * Cn + c) * HDn * En;  // [HD,E]
  const __bf16* Wk  = wk + (size_t)(h * Cn + c) * HDn * En;
  const __bf16* Wv  = wv + (size_t)(h * Cn + c) * HDn * En;

  v8f accQ[4], accK[4], accV[4];
#pragma unroll
  for (int t = 0; t < 4; ++t) { accQ[t] = v8f_zero(); accK[t] = v8f_zero(); accV[t] = v8f_zero(); }

  for (int k0 = 0; k0 < En; k0 += 32) {
    v16bf a = load_a_bf16(xbc, En, p0, k0, lane);
#pragma unroll
    for (int t = 0; t < 4; ++t) {
      accQ[t] = wmma_bf16(a, load_b_bf16(Wq, En, t * 16, k0, lane), accQ[t]);
      accK[t] = wmma_bf16(a, load_b_bf16(Wk, En, t * 16, k0, lane), accK[t]);
      accV[t] = wmma_bf16(a, load_b_bf16(Wv, En, t * 16, k0, lane), accV[t]);
    }
  }

  size_t base = (size_t)((b * Hn + h) * Cn + c) * Pn * HDn;
  int half = lane >> 4, n = lane & 15;
#pragma unroll
  for (int t = 0; t < 4; ++t) {
#pragma unroll
    for (int g = 0; g < 8; ++g) {
      int row = p0 + g + 8 * half;      // C/D layout: M = g + 8*(lane/16)
      int col = t * 16 + n;             // N = lane & 15
      Qb[base + (size_t)row * HDn + col] = (__bf16)accQ[t][g];
      Kb[base + (size_t)row * HDn + col] = (__bf16)accK[t][g];
      Vt[base + (size_t)col * Pn + row]  = (__bf16)accV[t][g];  // V stored transposed [HD,P]
    }
  }
}

// ======================= [2] flash attention =======================
// grid: B*H*C*(P/128)=1024 blocks x 256 thr; wave -> 16 query rows, loops 32-key blocks.
__global__ __launch_bounds__(256) void attn_kernel(
    const __bf16* __restrict__ Qb, const __bf16* __restrict__ Kb,
    const __bf16* __restrict__ Vt, __bf16* __restrict__ catb) {
  __shared__ __bf16 lds[8][16][32];  // per-wave 16x32 probs tile (1KB/wave)

  int blk = blockIdx.x;
  int ptile = blk & 7, h = (blk >> 3) & 7, c = (blk >> 6) & 3, b = blk >> 8;
  int wave = threadIdx.x >> 5, lane = threadIdx.x & 31;
  int half = lane >> 4, ln = lane & 15;
  int p0 = ptile * 128 + wave * 16;

  size_t base = (size_t)((b * Hn + h) * Cn + c) * Pn * HDn;
  const __bf16* Q = Qb + base;   // [P,HD]
  const __bf16* K = Kb + base;   // [P,HD]
  const __bf16* V = Vt + base;   // [HD,P] (transposed)

  // Q A-operands held in registers for the whole row-tile (K-dim = HD = 64 -> 2 chunks)
  v16bf qa0 = load_a_bf16(Q, HDn, p0, 0, lane);
  v16bf qa1 = load_a_bf16(Q, HDn, p0, 32, lane);

  float mrow[8], lrow[8];
  v8f o[4];
#pragma unroll
  for (int g = 0; g < 8; ++g) { mrow[g] = -3.0e38f; lrow[g] = 0.0f; }
#pragma unroll
  for (int t = 0; t < 4; ++t) o[t] = v8f_zero();

  const float scale = 0.125f;  // 1/sqrt(64)

  for (int j0 = 0; j0 < Pn; j0 += 32) {
    if (j0 + 32 < Pn) {  // prefetch next key/value block
      __builtin_prefetch(K + (size_t)(j0 + 32) * HDn, 0, 1);
      __builtin_prefetch(V + (j0 + 32), 0, 1);
    }
    // S[16x32] = Q[16x64] @ K_block[32x64]^T, two 16-key N-tiles
    v8f s0 = wmma_bf16(qa0, load_b_bf16(K, HDn, j0, 0, lane), v8f_zero());
    s0     = wmma_bf16(qa1, load_b_bf16(K, HDn, j0, 32, lane), s0);
    v8f s1 = wmma_bf16(qa0, load_b_bf16(K, HDn, j0 + 16, 0, lane), v8f_zero());
    s1     = wmma_bf16(qa1, load_b_bf16(K, HDn, j0 + 16, 32, lane), s1);

    // online softmax per row (row = g + 8*half; 16 lanes of a half hold one row)
#pragma unroll
    for (int g = 0; g < 8; ++g) {
      float a0 = s0[g] * scale, a1 = s1[g] * scale;
      float mx = fmaxf(a0, a1);
      mx = fmaxf(mx, __shfl_xor(mx, 1, 16));
      mx = fmaxf(mx, __shfl_xor(mx, 2, 16));
      mx = fmaxf(mx, __shfl_xor(mx, 4, 16));
      mx = fmaxf(mx, __shfl_xor(mx, 8, 16));
      float newm  = fmaxf(mrow[g], mx);
      float alpha = __expf(mrow[g] - newm);
      float e0 = __expf(a0 - newm), e1 = __expf(a1 - newm);
      float sum = e0 + e1;
      sum += __shfl_xor(sum, 1, 16);
      sum += __shfl_xor(sum, 2, 16);
      sum += __shfl_xor(sum, 4, 16);
      sum += __shfl_xor(sum, 8, 16);
      lrow[g] = lrow[g] * alpha + sum;
      mrow[g] = newm;
#pragma unroll
      for (int t = 0; t < 4; ++t) o[t][g] *= alpha;
      int r = g + 8 * half;
      lds[wave][r][ln]      = (__bf16)e0;   // D-layout -> row-major probs in LDS
      lds[wave][r][16 + ln] = (__bf16)e1;
    }
    asm volatile("" ::: "memory");  // keep DS store->load order (in-order per wave)

    // reload probs as A operand (16x32)
    v16bf pa;
    {
      const __bf16* lp = &lds[wave][ln][half * 8];
#pragma unroll
      for (int i = 0; i < 8; ++i) { pa[i] = lp[i]; pa[8 + i] = lp[16 + i]; }
    }
    asm volatile("" ::: "memory");

    // O[16x64] += P[16x32] @ V_block[32x64]  (B cols contiguous thanks to V^T)
#pragma unroll
    for (int t = 0; t < 4; ++t)
      o[t] = wmma_bf16(pa, load_b_bf16(V, Pn, t * 16, j0, lane), o[t]);
  }

  // finalize + write concatenated heads: cat[b,c,p, h*64 + d] (bf16)
  __bf16* cbc = catb + (size_t)(b * Cn + c) * Pn * En;
#pragma unroll
  for (int t = 0; t < 4; ++t) {
#pragma unroll
    for (int g = 0; g < 8; ++g) {
      int row = p0 + g + 8 * half;
      int col = h * HDn + t * 16 + ln;
      cbc[(size_t)row * En + col] = (__bf16)(o[t][g] / lrow[g]);
    }
  }
}

// ======================= [3] output projection =======================
// grid: B*C*(P/32)=512 blocks x 256 thr; wave -> 16 rows x 128 cols.
__global__ __launch_bounds__(256) void outp_kernel(
    const __bf16* __restrict__ catb, const __bf16* __restrict__ wo,
    float* __restrict__ out) {
  int blk = blockIdx.x;
  int ptile = blk & 31, c = (blk >> 5) & 3, b = blk >> 7;
  int wave = threadIdx.x >> 5, lane = threadIdx.x & 31;
  int msub = wave & 1, nq = wave >> 1;
  int p0 = ptile * 32 + msub * 16;
  int n0 = nq * 128;

  const __bf16* A = catb + (size_t)(b * Cn + c) * Pn * En;  // [P,E]
  const __bf16* W = wo + (size_t)c * En * En;               // [E,E] (row f holds e contiguous)

  v8f acc[8];
#pragma unroll
  for (int t = 0; t < 8; ++t) acc[t] = v8f_zero();

  for (int k0 = 0; k0 < En; k0 += 32) {
    v16bf a = load_a_bf16(A, En, p0, k0, lane);
#pragma unroll
    for (int t = 0; t < 8; ++t)
      acc[t] = wmma_bf16(a, load_b_bf16(W, En, n0 + t * 16, k0, lane), acc[t]);
  }

  int half = lane >> 4, n = lane & 15;
  float* obc = out + (size_t)(b * Cn + c) * Pn * En;
#pragma unroll
  for (int t = 0; t < 8; ++t) {
#pragma unroll
    for (int g = 0; g < 8; ++g) {
      int row = p0 + g + 8 * half;
      obc[(size_t)row * En + n0 + t * 16 + n] = acc[t][g];
    }
  }
}

// ======================= launch =======================
extern "C" void kernel_launch(void* const* d_in, const int* in_sizes, int n_in,
                              void* d_out, int out_size, void* d_ws, size_t ws_size,
                              hipStream_t stream) {
  const float* x  = (const float*)d_in[0];
  const float* wq = (const float*)d_in[1];
  const float* wk = (const float*)d_in[2];
  const float* wv = (const float*)d_in[3];
  const float* wo = (const float*)d_in[4];
  float* out = (float*)d_out;

  // workspace layout (bf16):
  //   Q | K | V^T | cat | x_bf16 | wq_bf16 | wk_bf16 | wv_bf16 | wo_bf16   (~92 MB)
  const size_t QSZ = (size_t)Bn * Hn * Cn * Pn * HDn;  // 8,388,608
  const size_t XSZ = (size_t)Bn * Cn * Pn * En;        // 8,388,608
  const size_t WSZ = (size_t)Hn * Cn * HDn * En;       // 1,048,576
  const size_t OSZ = (size_t)Cn * En * En;             // 1,048,576

  __bf16* Qb   = (__bf16*)d_ws;
  __bf16* Kb   = Qb + QSZ;
  __bf16* Vt   = Kb + QSZ;
  __bf16* catb = Vt + QSZ;
  __bf16* xb   = catb + XSZ;
  __bf16* wqb  = xb + XSZ;
  __bf16* wkb  = wqb + WSZ;
  __bf16* wvb  = wkb + WSZ;
  __bf16* wob  = wvb + WSZ;

  // [0] one-shot bf16 conversion (8 floats/thread)
  cvt_kernel<<<dim3((int)(XSZ / 8 / 256)), dim3(256), 0, stream>>>(x,  xb,  (int)(XSZ / 8));
  cvt_kernel<<<dim3((int)(WSZ / 8 / 256)), dim3(256), 0, stream>>>(wq, wqb, (int)(WSZ / 8));
  cvt_kernel<<<dim3((int)(WSZ / 8 / 256)), dim3(256), 0, stream>>>(wk, wkb, (int)(WSZ / 8));
  cvt_kernel<<<dim3((int)(WSZ / 8 / 256)), dim3(256), 0, stream>>>(wv, wvb, (int)(WSZ / 8));
  cvt_kernel<<<dim3((int)(OSZ / 8 / 256)), dim3(256), 0, stream>>>(wo, wob, (int)(OSZ / 8));

  qkv_kernel<<<dim3(1024), dim3(256), 0, stream>>>(xb, wqb, wkb, wvb, Qb, Kb, Vt);
  attn_kernel<<<dim3(1024), dim3(256), 0, stream>>>(Qb, Kb, Vt, catb);
  outp_kernel<<<dim3(512), dim3(256), 0, stream>>>(catb, wob, out);
}
